// Topic_Att_85985245266367
// MI455X (gfx1250) — matrix operand order repaired and checked by reading, not compiled, not run on hardware
//
#include <hip/hip_runtime.h>
#include <hip/hip_bf16.h>
#include <stdint.h>

// ---------------------------------------------------------------------------
// Model dimensions (fixed by the reference)
// ---------------------------------------------------------------------------
#define V_   50000
#define NI_  300
#define NIP_ 320     // NI padded to 32
#define H_   512
#define G4H_ 2048    // 4*H
#define ENC_ 512
#define K_   50
#define KP_  64      // K padded to 16
#define L_   128
#define B_   32
#define S_   512
#define N_   262144
#define VK_  50016   // V padded to 32
#define BS_  16384   // B*S

typedef __attribute__((ext_vector_type(16))) __bf16 v16bf;
typedef __attribute__((ext_vector_type(8)))  float  v8f;
typedef __attribute__((ext_vector_type(4)))  unsigned int uint4_;

union Frag32 { uint4_ q[2]; v16bf v; };   // 32 bytes: one WMMA bf16 operand per lane

__device__ __forceinline__ float sigf(float x) { return 1.f / (1.f + __expf(-x)); }

// ---------------------------------------------------------------------------
// Generic bf16 WMMA GEMM:  C[M,N] = act(A[M,K] @ W[N,K]^T + bias)
// One 16x16 output tile per wave, K-loop in steps of 32 via
// v_wmma_f32_16x16x32_bf16 (f32 accumulate). M%16==0, N%16==0, K%32==0.
// store_bf16: write C as bf16 (for the LSTM input projections).
// ---------------------------------------------------------------------------
__global__ void gemm_bf16_kernel(const __bf16* __restrict__ A,
                                 const __bf16* __restrict__ W,
                                 const float*  __restrict__ bias,
                                 void*         __restrict__ C,
                                 int M, int N, int K, int act, int store_bf16) {
  const int wave = blockIdx.x * (blockDim.x >> 5) + (threadIdx.x >> 5);
  const int lane = threadIdx.x & 31;
  const int TN = N >> 4, TM = M >> 4;
  const int tm = wave / TN, tn = wave % TN;
  if (tm >= TM) return;                     // wave-uniform: EXEC stays full for WMMA

  const int m15 = lane & 15, kh = lane >> 4;
  const __bf16* pa = A + (size_t)(tm * 16 + m15) * K + kh * 8;
  const __bf16* pb = W + (size_t)(tn * 16 + m15) * K + kh * 8;

  v8f acc = {};
  for (int k0 = 0; k0 < K; k0 += 32) {
    Frag32 fa, fb;
    fa.q[0] = *(const uint4_*)(pa);
    fa.q[1] = *(const uint4_*)(pa + 16);
    fb.q[0] = *(const uint4_*)(pb);
    fb.q[1] = *(const uint4_*)(pb + 16);
    acc = __builtin_amdgcn_wmma_f32_16x16x32_bf16(false, fa.v, false, fb.v,
                                                  (short)0, acc, false, false);
    pa += 32; pb += 32;
  }

  const int cm = tm * 16 + kh * 8;
  const int cn = tn * 16 + m15;
  const float bv = bias ? bias[cn] : 0.f;
  if (store_bf16) {
    __bf16* Cb = (__bf16*)C;
    #pragma unroll
    for (int r = 0; r < 8; ++r) {
      float v = acc[r] + bv;
      if (act == 1) v = tanhf(v);
      Cb[(size_t)(cm + r) * N + cn] = (__bf16)v;
    }
  } else {
    float* Cf = (float*)C;
    #pragma unroll
    for (int r = 0; r < 8; ++r) {
      float v = acc[r] + bv;
      if (act == 1) v = tanhf(v);
      Cf[(size_t)(cm + r) * N + cn] = v;
    }
  }
}

// ---------------------------------------------------------------------------
// LSTM recurrence: one workgroup (32 wave32 waves) per direction.
//  - h (bf16) LDS-resident, c entirely in registers (static ownership map)
//  - per-step 32x2048x512 GEMM on v_wmma_f32_16x16x32_bf16
//  - xp gate biases (bf16, 128KB/step) double-buffered in LDS and prefetched
//    one step ahead with global_load_async_to_lds_b128 (ASYNCcnt), hiding the
//    seed-fetch latency behind the WMMA K-loop.
// LDS: h 32KB @0, xp buf0 128KB @32768, xp buf1 128KB @163840 -> 288KB total.
// ---------------------------------------------------------------------------
#define LDS_H_BYTES   (B_ * H_ * 2)            // 32768
#define LDS_XP_BYTES  (B_ * G4H_ * 2)          // 131072
#define LDS_XP0_OFF   LDS_H_BYTES              // 32768
#define LDS_XP1_OFF   (LDS_H_BYTES + LDS_XP_BYTES)

__global__ void lstm_kernel(const __bf16* __restrict__ xpF,
                            const __bf16* __restrict__ xpR,
                            const __bf16* __restrict__ WhhF,
                            const __bf16* __restrict__ WhhR,
                            float* __restrict__ values) {
  const int dir = blockIdx.x;
  const __bf16* xp  = dir ? xpR  : xpF;
  const __bf16* Whh = dir ? WhhR : WhhF;

  extern __shared__ char smem[];
  __bf16* h_bf = (__bf16*)smem;   // [32][512]

  for (int i = threadIdx.x; i < B_ * H_; i += blockDim.x) h_bf[i] = (__bf16)0.f;

  // synchronous preload of step 0's xp slab into LDS buffer 0
  {
    const int ss0 = dir ? (S_ - 1) : 0;
    const uint4_* src = (const uint4_*)(xp + (size_t)ss0 * B_ * G4H_);
    uint4_* dst = (uint4_*)(smem + LDS_XP0_OFF);
    for (int i = threadIdx.x; i < (B_ * G4H_) / 8; i += blockDim.x) dst[i] = src[i];
  }
  __syncthreads();

  const int lane = threadIdx.x & 31;
  const int w    = threadIdx.x >> 5;     // 0..31
  const int r    = w & 1;                // row tile (batch 0..15 / 16..31)
  const int nc0  = w >> 1;               // 0..15
  const int m15  = lane & 15, kh = lane >> 4;
  const unsigned tid16 = (unsigned)threadIdx.x * 16u;

  float creg[2][8];                      // cell state lives in registers
  #pragma unroll
  for (int hh = 0; hh < 2; ++hh)
    #pragma unroll
    for (int rr = 0; rr < 8; ++rr) creg[hh][rr] = 0.f;

  for (int s = 0; s < S_; ++s) {
    const int ss  = dir ? (S_ - 1 - s) : s;
    const int cur = s & 1, nxt = cur ^ 1;

    // async prefetch of next step's xp slab into the alternate LDS buffer
    if (s + 1 < S_) {
      const int ssn = dir ? (S_ - 2 - s) : (s + 1);
      const char* srcBase = (const char*)(xp + (size_t)ssn * B_ * G4H_);
      const unsigned ldsBase = (unsigned)(LDS_XP0_OFF + nxt * LDS_XP_BYTES) + tid16;
      #pragma unroll
      for (int it = 0; it < 8; ++it) {
        unsigned loff = ldsBase + (unsigned)(it * 16384);
        unsigned long long ga =
            (unsigned long long)(uintptr_t)(srcBase + tid16 + it * 16384);
        asm volatile("global_load_async_to_lds_b128 %0, %1, off"
                     :: "v"(loff), "v"(ga) : "memory");
      }
    }

    const __bf16* xpb = (const __bf16*)(smem + LDS_XP0_OFF + cur * LDS_XP_BYTES);

    v8f acc[2][4];
    for (int half = 0; half < 2; ++half) {
      const int ncol = nc0 + half * 16;               // n tile in [0,512)
      const int cm = r * 16 + kh * 8;
      // seed accumulators with xp from LDS (includes bias from input-proj GEMM)
      for (int g = 0; g < 4; ++g) {
        const int n = g * H_ + ncol * 16 + m15;
        #pragma unroll
        for (int rr = 0; rr < 8; ++rr)
          acc[half][g][rr] = (float)xpb[(cm + rr) * G4H_ + n];
      }
      // h_prev @ Whh^T for the 4 gate tiles at this n offset
      for (int k0 = 0; k0 < H_; k0 += 32) {
        Frag32 fa;
        const __bf16* pa = h_bf + (r * 16 + m15) * H_ + k0 + kh * 8;
        fa.q[0] = *(const uint4_*)(pa);
        fa.q[1] = *(const uint4_*)(pa + 16);
        #pragma unroll
        for (int g = 0; g < 4; ++g) {
          const int nrow = g * H_ + ncol * 16 + m15;
          Frag32 fb;
          const __bf16* pb = Whh + (size_t)nrow * H_ + k0 + kh * 8;
          fb.q[0] = *(const uint4_*)(pb);
          fb.q[1] = *(const uint4_*)(pb + 16);
          acc[half][g] = __builtin_amdgcn_wmma_f32_16x16x32_bf16(
              false, fa.v, false, fb.v, (short)0, acc[half][g], false, false);
        }
      }
    }
    __syncthreads();   // all h reads done before h is overwritten

    for (int half = 0; half < 2; ++half) {
      const int ncol = nc0 + half * 16;
      const int n = ncol * 16 + m15;
      const int bbase = r * 16 + kh * 8;
      #pragma unroll
      for (int rr = 0; rr < 8; ++rr) {
        const int b = bbase + rr;
        const float gi = acc[half][0][rr], gf = acc[half][1][rr];
        const float gg = acc[half][2][rr], go = acc[half][3][rr];
        const float cNew = sigf(gf) * creg[half][rr] + sigf(gi) * tanhf(gg);
        const float hNew = sigf(go) * tanhf(cNew);
        creg[half][rr] = cNew;
        h_bf[b * H_ + n] = (__bf16)hNew;
        values[((size_t)b * S_ + ss) * (2 * H_) + dir * H_ + n] = hNew;
      }
    }
    asm volatile("s_wait_asynccnt 0x0" ::: "memory");  // prefetch landed
    __syncthreads();   // h + xp buffer visible to all before next step
  }
}

// ---------------------------------------------------------------------------
// Small utility / elementwise kernels
// ---------------------------------------------------------------------------
__global__ void zero_f32_kernel(float* p, int n) {
  int i = blockIdx.x * blockDim.x + threadIdx.x;
  if (i < n) p[i] = 0.f;
}

__global__ void scatter_bow_kernel(const int* __restrict__ text,
                                   const int* __restrict__ text_batch,
                                   const float* __restrict__ text_w,
                                   float* __restrict__ bow, int n) {
  int i = blockIdx.x * blockDim.x + threadIdx.x;
  if (i < n) atomicAdd(&bow[(size_t)text_batch[i] * V_ + text[i]], text_w[i]);
}

// convert f32 -> bf16 with row and column zero-padding
__global__ void cvt_pad_bf16_kernel(const float* __restrict__ src,
                                    __bf16* __restrict__ dst,
                                    int srows, int drows, int K, int Kp) {
  long long idx = (long long)blockIdx.x * blockDim.x + threadIdx.x;
  long long total = (long long)drows * Kp;
  if (idx >= total) return;
  int rr = (int)(idx / Kp), c = (int)(idx % Kp);
  float v = (rr < srows && c < K) ? src[(size_t)rr * K + c] : 0.f;
  dst[idx] = (__bf16)v;
}

__global__ void pad_f32_kernel(const float* __restrict__ src, float* dst, int n, int np) {
  int i = blockIdx.x * blockDim.x + threadIdx.x;
  if (i < np) dst[i] = (i < n) ? src[i] : 0.f;
}

// emb rows ordered m = s*B + b (matches xp GEMM output layout [s][b][4H])
__global__ void emb_gather_kernel(const int* __restrict__ x,
                                  const __bf16* __restrict__ wv_bf,
                                  __bf16* __restrict__ emb) {
  long long idx = (long long)blockIdx.x * blockDim.x + threadIdx.x;
  const long long total = (long long)BS_ * NIP_;
  if (idx >= total) return;
  int m = (int)(idx / NIP_), c = (int)(idx % NIP_);
  int s = m >> 5, b = m & 31;
  int tok = x[b * S_ + s];
  emb[idx] = wv_bf[(size_t)tok * NIP_ + c];
}

// theta = softmax(mean @ dec_W^T + dec_b) ; one block per batch row
__global__ void theta_kernel(const float* __restrict__ meanO,  // [B, KP]
                             const float* __restrict__ dec_W,  // [K, K]
                             const float* __restrict__ dec_b,
                             float* __restrict__ theta) {      // [B, KP]
  int b = blockIdx.x, t = threadIdx.x;
  __shared__ float sv[KP_];
  __shared__ float mx, sum;
  float v = -1e30f;
  if (t < K_) {
    float s = dec_b[t];
    for (int j = 0; j < K_; ++j) s += meanO[b * KP_ + j] * dec_W[t * K_ + j];
    v = s;
  }
  sv[t] = v;
  __syncthreads();
  if (t == 0) {
    float m = -1e30f;
    for (int j = 0; j < K_; ++j) m = fmaxf(m, sv[j]);
    float ss = 0.f;
    for (int j = 0; j < K_; ++j) ss += __expf(sv[j] - m);
    mx = m; sum = ss;
  }
  __syncthreads();
  theta[b * KP_ + t] = (t < K_) ? __expf(sv[t] - mx) / sum : 0.f;
}

// in-place softmax along a row (beta: rows of length 50000)
__global__ void row_softmax_kernel(float* data, int cols) {
  float* p = data + (size_t)blockIdx.x * cols;
  __shared__ float red[256];
  int t = threadIdx.x;
  float m = -1e30f;
  for (int i = t; i < cols; i += 256) m = fmaxf(m, p[i]);
  red[t] = m; __syncthreads();
  for (int o = 128; o > 0; o >>= 1) { if (t < o) red[t] = fmaxf(red[t], red[t + o]); __syncthreads(); }
  m = red[0]; __syncthreads();
  float s = 0.f;
  for (int i = t; i < cols; i += 256) s += __expf(p[i] - m);
  red[t] = s; __syncthreads();
  for (int o = 128; o > 0; o >>= 1) { if (t < o) red[t] += red[t + o]; __syncthreads(); }
  s = red[0];
  float inv = 1.f / s;
  for (int i = t; i < cols; i += 256) p[i] = __expf(p[i] - m) * inv;
}

// NL[b] += -sum_v bow[b,v]*log(recon[b,v]+1e-10), recon = theta @ beta
__global__ void nl_kernel(const float* __restrict__ bow,
                          const float* __restrict__ theta,   // [B,KP]
                          const float* __restrict__ beta,    // [KP,V] rows<K valid
                          float* __restrict__ NL) {
  int b = blockIdx.y, t = threadIdx.x;
  __shared__ float th[K_];
  __shared__ float red[256];
  if (t < K_) th[t] = theta[b * KP_ + t];
  __syncthreads();
  int v = blockIdx.x * 256 + t;
  float part = 0.f;
  if (v < V_) {
    float recon = 0.f;
    for (int k = 0; k < K_; ++k) recon += th[k] * beta[(size_t)k * V_ + v];
    part = bow[(size_t)b * V_ + v] * __logf(recon + 1e-10f);
  }
  red[t] = part; __syncthreads();
  for (int o = 128; o > 0; o >>= 1) { if (t < o) red[t] += red[t + o]; __syncthreads(); }
  if (t == 0) atomicAdd(&NL[b], -red[0]);
}

// topic_loss = mean_b( NL[b] + KLD[b] )
__global__ void kld_loss_kernel(const float* __restrict__ meanO,
                                const float* __restrict__ logvarO,
                                const float* __restrict__ NL,
                                float* __restrict__ topic_loss) {
  __shared__ float red[B_];
  int b = threadIdx.x;
  float kld = 0.f;
  for (int k = 0; k < K_; ++k) {
    float m = meanO[b * KP_ + k], lv = logvarO[b * KP_ + k];
    kld += 1.f - m * m + lv - __expf(lv);
  }
  kld *= -0.5f;
  red[b] = NL[b] + kld;
  __syncthreads();
  if (b == 0) {
    float s = 0.f;
    for (int i = 0; i < B_; ++i) s += red[i];
    *topic_loss = s / (float)B_;
  }
}

// softmax of scores over the sequence axis, per (b, k<K), in-place
__global__ void attw_softmax_kernel(float* __restrict__ scores) {
  int idx = blockIdx.x * blockDim.x + threadIdx.x;
  if (idx >= B_ * KP_) return;
  int b = idx >> 6, k = idx & 63;
  if (k >= K_) return;
  float* p = scores + (size_t)b * S_ * KP_ + k;
  float m = -1e30f;
  for (int s = 0; s < S_; ++s) m = fmaxf(m, p[(size_t)s * KP_]);
  float sum = 0.f;
  for (int s = 0; s < S_; ++s) sum += __expf(p[(size_t)s * KP_] - m);
  float inv = 1.f / sum;
  for (int s = 0; s < S_; ++s) p[(size_t)s * KP_] = __expf(p[(size_t)s * KP_] - m) * inv;
}

__global__ void topic_atten_kernel(const float* __restrict__ attw,   // [B*S, KP]
                                   const float* __restrict__ theta,  // [B, KP]
                                   float* __restrict__ ta) {         // [B, S]
  int idx = blockIdx.x * blockDim.x + threadIdx.x;
  if (idx >= B_ * S_) return;
  int b = idx >> 9;
  const float* p = attw + (size_t)idx * KP_;
  float acc = 0.f;
  for (int k = 0; k < K_; ++k) acc += p[k] * theta[b * KP_ + k];
  ta[idx] = acc;
}

__global__ void atten_out_kernel(const float* __restrict__ ta,      // [B, S]
                                 const float* __restrict__ values,  // [B, S, 2H]
                                 float* __restrict__ ao) {          // [B, 2H]
  int idx = blockIdx.x * blockDim.x + threadIdx.x;
  if (idx >= B_ * 2 * H_) return;
  int b = idx >> 10, n = idx & 1023;
  float acc = 0.f;
  for (int s = 0; s < S_; ++s)
    acc += ta[b * S_ + s] * values[((size_t)b * S_ + s) * (2 * H_) + n];
  ao[idx] = acc;
}

__global__ void fc_kernel(const float* __restrict__ ao,
                          const float* __restrict__ fc_W,
                          const float* __restrict__ fc_b,
                          float* __restrict__ out) {
  int idx = blockIdx.x * blockDim.x + threadIdx.x;
  if (idx >= B_ * L_) return;
  int b = idx >> 7, l = idx & 127;
  float acc = fc_b[l];
  for (int j = 0; j < 2 * H_; ++j) acc += ao[b * 2 * H_ + j] * fc_W[l * 2 * H_ + j];
  out[idx] = 1.f / (1.f + __expf(-acc));
}

__global__ void bce_final_kernel(const float* __restrict__ out,
                                 const float* __restrict__ target,
                                 const float* __restrict__ topic_loss,
                                 float* __restrict__ loss_out) {
  __shared__ float red[1024];
  int t = threadIdx.x;
  float acc = 0.f;
  for (int i = t; i < B_ * L_; i += 1024) {
    float p = fminf(fmaxf(out[i], 1e-7f), 1.f - 1e-7f);
    float tg = target[i];
    acc -= tg * __logf(p) + (1.f - tg) * __logf(1.f - p);
  }
  red[t] = acc; __syncthreads();
  for (int o = 512; o > 0; o >>= 1) { if (t < o) red[t] += red[t + o]; __syncthreads(); }
  if (t == 0) *loss_out = red[0] / (float)(B_ * L_) + *topic_loss;
}

// ---------------------------------------------------------------------------
// Host-side orchestration
// ---------------------------------------------------------------------------
extern "C" void kernel_launch(void* const* d_in, const int* in_sizes, int n_in,
                              void* d_out, int out_size, void* d_ws, size_t ws_size,
                              hipStream_t stream) {
  const int*   x          = (const int*)d_in[0];
  const int*   text       = (const int*)d_in[1];
  const int*   text_batch = (const int*)d_in[2];
  const float* text_w     = (const float*)d_in[3];
  const float* target     = (const float*)d_in[4];
  const float* word_vec   = (const float*)d_in[5];
  const float* topic_vec  = (const float*)d_in[6];
  const float* enc1_W     = (const float*)d_in[7];
  const float* enc1_b     = (const float*)d_in[8];
  const float* enc2_W     = (const float*)d_in[9];
  const float* enc2_b     = (const float*)d_in[10];
  const float* mean_W     = (const float*)d_in[11];
  const float* mean_b     = (const float*)d_in[12];
  const float* logvar_W   = (const float*)d_in[13];
  const float* logvar_b   = (const float*)d_in[14];
  const float* dec_W      = (const float*)d_in[15];
  const float* dec_b      = (const float*)d_in[16];
  const float* Wih_f      = (const float*)d_in[17];
  const float* Whh_f      = (const float*)d_in[18];
  const float* b_f        = (const float*)d_in[19];
  const float* Wih_r      = (const float*)d_in[20];
  const float* Whh_r      = (const float*)d_in[21];
  const float* b_r        = (const float*)d_in[22];
  const float* att_W      = (const float*)d_in[23];
  const float* att_b      = (const float*)d_in[24];
  const float* fc_W       = (const float*)d_in[25];
  const float* fc_b       = (const float*)d_in[26];

  float* out = (float*)d_out;
  float* loss_ptr = out + (out_size - 1);

  // workspace arena
  size_t off = 0;
  auto alloc = [&](size_t bytes) -> char* {
    char* p = (char*)d_ws + off;
    off = (off + bytes + 255) & ~(size_t)255;
    return p;
  };
  float*  bow        = (float*)alloc((size_t)B_ * V_ * 4);
  __bf16* bow_bf     = (__bf16*)alloc((size_t)B_ * VK_ * 2);
  __bf16* enc1W_bf   = (__bf16*)alloc((size_t)1024 * VK_ * 2);
  float*  enc1o      = (float*)alloc((size_t)B_ * 1024 * 4);
  __bf16* enc1_bf    = (__bf16*)alloc((size_t)B_ * 1024 * 2);
  __bf16* enc2W_bf   = (__bf16*)alloc((size_t)ENC_ * 1024 * 2);
  float*  enc2o      = (float*)alloc((size_t)B_ * ENC_ * 4);
  __bf16* enc2_bf    = (__bf16*)alloc((size_t)B_ * ENC_ * 2);
  __bf16* meanW_bf   = (__bf16*)alloc((size_t)KP_ * ENC_ * 2);
  __bf16* logvarW_bf = (__bf16*)alloc((size_t)KP_ * ENC_ * 2);
  float*  mean_b_p   = (float*)alloc(KP_ * 4);
  float*  logvar_b_p = (float*)alloc(KP_ * 4);
  float*  meanO      = (float*)alloc((size_t)B_ * KP_ * 4);
  float*  logvarO    = (float*)alloc((size_t)B_ * KP_ * 4);
  float*  theta      = (float*)alloc((size_t)B_ * KP_ * 4);
  __bf16* topicv_bf  = (__bf16*)alloc((size_t)KP_ * NIP_ * 2);
  __bf16* wordv_bf   = (__bf16*)alloc((size_t)V_ * NIP_ * 2);
  float*  beta       = (float*)alloc((size_t)KP_ * V_ * 4);
  float*  NL         = (float*)alloc(B_ * 4);
  float*  topic_loss = (float*)alloc(4);
  __bf16* emb_bf     = (__bf16*)alloc((size_t)BS_ * NIP_ * 2);
  __bf16* WihF_bf    = (__bf16*)alloc((size_t)G4H_ * NIP_ * 2);
  __bf16* WihR_bf    = (__bf16*)alloc((size_t)G4H_ * NIP_ * 2);
  __bf16* WhhF_bf    = (__bf16*)alloc((size_t)G4H_ * H_ * 2);
  __bf16* WhhR_bf    = (__bf16*)alloc((size_t)G4H_ * H_ * 2);
  __bf16* xpF        = (__bf16*)alloc((size_t)BS_ * G4H_ * 2);   // bf16 now
  __bf16* xpR        = (__bf16*)alloc((size_t)BS_ * G4H_ * 2);
  float*  values     = (float*)alloc((size_t)BS_ * 2 * H_ * 4);
  __bf16* values_bf  = (__bf16*)alloc((size_t)BS_ * 2 * H_ * 2);
  __bf16* attW_bf    = (__bf16*)alloc((size_t)304 * 2 * H_ * 2);
  float*  att_b_p    = (float*)alloc(304 * 4);
  float*  h1         = (float*)alloc((size_t)BS_ * 304 * 4);
  __bf16* h1_bf      = (__bf16*)alloc((size_t)BS_ * NIP_ * 2);
  float*  scores     = (float*)alloc((size_t)BS_ * KP_ * 4);
  float*  ta         = (float*)alloc((size_t)B_ * S_ * 4);
  float*  ao         = (float*)alloc((size_t)B_ * 2 * H_ * 4);
  (void)ws_size; (void)n_in; (void)in_sizes;

  auto cvt = [&](const float* src, __bf16* dst, int sr, int dr, int K, int Kp) {
    long long total = (long long)dr * Kp;
    int blocks = (int)((total + 255) / 256);
    cvt_pad_bf16_kernel<<<blocks, 256, 0, stream>>>(src, dst, sr, dr, K, Kp);
  };
  auto gemm = [&](const __bf16* A, const __bf16* W, const float* bias, void* C,
                  int M, int N, int K, int act, int sbf) {
    int waves = (M / 16) * (N / 16);
    int blocks = (waves + 7) / 8;
    gemm_bf16_kernel<<<blocks, 256, 0, stream>>>(A, W, bias, C, M, N, K, act, sbf);
  };

  // ---- topic model path ----
  zero_f32_kernel<<<(B_ * V_ + 255) / 256, 256, 0, stream>>>(bow, B_ * V_);
  zero_f32_kernel<<<1, 256, 0, stream>>>(NL, B_);
  scatter_bow_kernel<<<(N_ + 255) / 256, 256, 0, stream>>>(text, text_batch, text_w, bow, N_);

  cvt(bow, bow_bf, B_, B_, V_, VK_);
  cvt(enc1_W, enc1W_bf, 1024, 1024, V_, VK_);
  gemm(bow_bf, enc1W_bf, enc1_b, enc1o, B_, 1024, VK_, 1, 0);       // tanh
  cvt(enc1o, enc1_bf, B_, B_, 1024, 1024);
  cvt(enc2_W, enc2W_bf, ENC_, ENC_, 1024, 1024);
  gemm(enc1_bf, enc2W_bf, enc2_b, enc2o, B_, ENC_, 1024, 1, 0);     // tanh
  cvt(enc2o, enc2_bf, B_, B_, ENC_, ENC_);
  cvt(mean_W, meanW_bf, K_, KP_, ENC_, ENC_);
  cvt(logvar_W, logvarW_bf, K_, KP_, ENC_, ENC_);
  pad_f32_kernel<<<1, KP_, 0, stream>>>(mean_b, mean_b_p, K_, KP_);
  pad_f32_kernel<<<1, KP_, 0, stream>>>(logvar_b, logvar_b_p, K_, KP_);
  gemm(enc2_bf, meanW_bf, mean_b_p, meanO, B_, KP_, ENC_, 0, 0);
  gemm(enc2_bf, logvarW_bf, logvar_b_p, logvarO, B_, KP_, ENC_, 0, 0);
  theta_kernel<<<B_, KP_, 0, stream>>>(meanO, dec_W, dec_b, theta);

  cvt(topic_vec, topicv_bf, K_, KP_, NI_, NIP_);
  cvt(word_vec, wordv_bf, V_, V_, NI_, NIP_);
  gemm(topicv_bf, wordv_bf, nullptr, beta, KP_, V_, NIP_, 0, 0);
  row_softmax_kernel<<<K_, 256, 0, stream>>>(beta, V_);
  nl_kernel<<<dim3((V_ + 255) / 256, B_), 256, 0, stream>>>(bow, theta, beta, NL);

  // ---- LSTM path ----
  emb_gather_kernel<<<(int)(((long long)BS_ * NIP_ + 255) / 256), 256, 0, stream>>>(x, wordv_bf, emb_bf);
  cvt(Wih_f, WihF_bf, G4H_, G4H_, NI_, NIP_);
  cvt(Wih_r, WihR_bf, G4H_, G4H_, NI_, NIP_);
  cvt(Whh_f, WhhF_bf, G4H_, G4H_, H_, H_);
  cvt(Whh_r, WhhR_bf, G4H_, G4H_, H_, H_);
  gemm(emb_bf, WihF_bf, b_f, xpF, BS_, G4H_, NIP_, 0, 1);           // bf16 out
  gemm(emb_bf, WihR_bf, b_r, xpR, BS_, G4H_, NIP_, 0, 1);           // bf16 out

  const size_t lstm_lds = LDS_XP1_OFF + LDS_XP_BYTES;               // 288 KB
  lstm_kernel<<<2, 1024, lstm_lds, stream>>>(xpF, xpR, WhhF_bf, WhhR_bf, values);

  // ---- attention path ----
  cvt(values, values_bf, BS_, BS_, 2 * H_, 2 * H_);
  cvt(att_W, attW_bf, NI_, 304, 2 * H_, 2 * H_);
  pad_f32_kernel<<<2, 256, 0, stream>>>(att_b, att_b_p, NI_, 304);
  gemm(values_bf, attW_bf, att_b_p, h1, BS_, 304, 2 * H_, 1, 0);    // tanh
  cvt(h1, h1_bf, BS_, BS_, 304, NIP_);
  gemm(h1_bf, topicv_bf, nullptr, scores, BS_, KP_, NIP_, 0, 0);
  attw_softmax_kernel<<<(B_ * KP_ + 255) / 256, 256, 0, stream>>>(scores);
  topic_atten_kernel<<<(B_ * S_ + 255) / 256, 256, 0, stream>>>(scores, theta, ta);
  atten_out_kernel<<<(B_ * 2 * H_ + 255) / 256, 256, 0, stream>>>(ta, values, ao);

  // ---- losses / output ----
  kld_loss_kernel<<<1, B_, 0, stream>>>(meanO, logvarO, NL, topic_loss);
  fc_kernel<<<(B_ * L_ + 255) / 256, 256, 0, stream>>>(ao, fc_W, fc_b, out);
  bce_final_kernel<<<1, 1024, 0, stream>>>(out, target, topic_loss, loss_ptr);
}
